// Seq2Seq_56375740727620
// MI455X (gfx1250) — matrix operand order repaired and checked by reading, not compiled, not run on hardware
//
#include <hip/hip_runtime.h>
#include <hip/hip_bf16.h>
#include <math.h>

typedef __attribute__((ext_vector_type(16))) __bf16 v16bf;
typedef __attribute__((ext_vector_type(8)))  __bf16 v8bf;
typedef __attribute__((ext_vector_type(8)))  float  v8f;

#define B_  32
#define S_  128
#define T_  128
#define V_  32000
#define E_  512
#define H_  1024
#define H3_ 3072
#define M_ENC (B_ * S_)        // 4096
#define M_DEC (B_ * (T_ - 1))  // 4064

// ---------------------------------------------------------------------------
// WMMA tile loaders (bf16, 16x16x32; contiguous k order k = 16*(L/16) + s, same in A and B)
// A (16x32, MxK): lane L, M=L%16; lanes 0-15 hold K kk..kk+15;
//                 lanes 16-31 hold K kk+16..kk+31.
// B (32x16, KxN): lane L, N=L%16; lanes 0-15 hold K kk..kk+15; 16-31 kk+16..kk+31.
// C/D (16x16 f32): lane L, N=L%16; VGPR i holds M = i + (L>=16 ? 8 : 0).
// ---------------------------------------------------------------------------
__device__ __forceinline__ v16bf cat8(v8bf lo, v8bf hi) {
  return __builtin_shufflevector(lo, hi, 0, 1, 2, 3, 4, 5, 6, 7,
                                 8, 9, 10, 11, 12, 13, 14, 15);
}

__device__ __forceinline__ v16bf load_a_tile(const __bf16* __restrict__ X, int ld,
                                             int mrow, int kk, int lane) {
  int kA = kk + ((lane >> 4) << 4);
  const __bf16* p = X + (size_t)mrow * ld + kA;
  v8bf lo = *(const v8bf*)(p);
  v8bf hi = *(const v8bf*)(p + 8);
  return cat8(lo, hi);
}

__device__ __forceinline__ v16bf load_b_tile(const __bf16* __restrict__ W, int ld,
                                             int nrow, int kk, int lane) {
  int kB = kk + ((lane >> 4) << 4);
  const __bf16* p = W + (size_t)nrow * ld + kB;
  v8bf lo = *(const v8bf*)(p);
  v8bf hi = *(const v8bf*)(p + 8);
  return cat8(lo, hi);
}

#define WMMA_BF16(a, b, c) \
  __builtin_amdgcn_wmma_f32_16x16x32_bf16(false, (a), false, (b), (short)0, (c), false, false)

// ---------------------------------------------------------------------------
// fp32 -> bf16 weight conversion (grid-stride)
// ---------------------------------------------------------------------------
__global__ void cvt_bf16_kernel(const float* __restrict__ in, __bf16* __restrict__ out,
                                size_t n) {
  for (size_t i = (size_t)blockIdx.x * blockDim.x + threadIdx.x; i < n;
       i += (size_t)gridDim.x * blockDim.x)
    out[i] = (__bf16)in[i];
}

// ---------------------------------------------------------------------------
// Embedding gather -> bf16, PAD token (0) row forced to zero.
// Row r = b*L + t ; token = toks[b*tokStride + t]
// ---------------------------------------------------------------------------
__global__ void gather_embed_kernel(const int* __restrict__ toks, int tokStride, int L,
                                    const float* __restrict__ emb,
                                    __bf16* __restrict__ X, int nrows) {
  size_t idx = (size_t)blockIdx.x * blockDim.x + threadIdx.x;
  size_t total = (size_t)nrows * E_;
  if (idx >= total) return;
  int row = (int)(idx / E_);
  int e = (int)(idx % E_);
  int b = row / L, t = row - b * L;
  int tok = toks[(size_t)b * tokStride + t];
  float v = (tok == 0) ? 0.0f : emb[(size_t)tok * E_ + e];
  X[idx] = (__bf16)v;
}

__global__ void zero_h_kernel(float* __restrict__ hf, __bf16* __restrict__ hb, int n) {
  int i = blockIdx.x * blockDim.x + threadIdx.x;
  if (i < n) { hf[i] = 0.0f; hb[i] = (__bf16)0.0f; }
}

__global__ void zero_t0_kernel(float* __restrict__ out) {
  int i = blockIdx.x * blockDim.x + threadIdx.x;
  if (i >= B_ * V_) return;
  int b = i / V_, v = i - b * V_;
  out[(size_t)b * T_ * V_ + v] = 0.0f;
}

// ---------------------------------------------------------------------------
// gx = X[M,K] @ W[N,K]^T + bias[N]   (bf16 WMMA, f32 out, row-major ldo)
// grid: (M/16, N/64), block: 32 (one wave -> 16x64 tile, 4 accumulators)
// ---------------------------------------------------------------------------
__device__ __forceinline__ void store_tile_bias(v8f c, int n, int mi,
                                                const float* __restrict__ bias,
                                                float* __restrict__ out, int ldo) {
  float bv = bias[n];
#pragma unroll
  for (int i = 0; i < 8; ++i)
    out[(size_t)(mi + i) * ldo + n] = c[i] + bv;
}

__global__ __launch_bounds__(32)
void gemm_xwT_bias_kernel(const __bf16* __restrict__ X, const __bf16* __restrict__ W,
                          const float* __restrict__ bias, float* __restrict__ out,
                          int K, int ldo) {
  const int lane = threadIdx.x;
  const int mbase = blockIdx.x * 16;
  const int nbase = blockIdx.y * 64;
  const int mrow = mbase + (lane & 15);
  const int nr = nbase + (lane & 15);
  v8f c0 = {}, c1 = {}, c2 = {}, c3 = {};
  for (int kk = 0; kk < K; kk += 32) {
    v16bf a = load_a_tile(X, K, mrow, kk, lane);
    v16bf b0 = load_b_tile(W, K, nr, kk, lane);
    v16bf b1 = load_b_tile(W, K, nr + 16, kk, lane);
    v16bf b2 = load_b_tile(W, K, nr + 32, kk, lane);
    v16bf b3 = load_b_tile(W, K, nr + 48, kk, lane);
    c0 = WMMA_BF16(a, b0, c0);
    c1 = WMMA_BF16(a, b1, c1);
    c2 = WMMA_BF16(a, b2, c2);
    c3 = WMMA_BF16(a, b3, c3);
  }
  const int mi = mbase + ((lane >> 4) << 3);
  store_tile_bias(c0, nbase + 0 + (lane & 15), mi, bias, out, ldo);
  store_tile_bias(c1, nbase + 16 + (lane & 15), mi, bias, out, ldo);
  store_tile_bias(c2, nbase + 32 + (lane & 15), mi, bias, out, ldo);
  store_tile_bias(c3, nbase + 48 + (lane & 15), mi, bias, out, ldo);
}

// ---------------------------------------------------------------------------
// Fused GRU step: gh = h_prev @ Whh^T (+bhh), gating, h_new write-out.
// Wave owns 16 batch rows x 16 hidden cols; computes r/z/n columns together
// so the gate nonlinearity is local to the wave. One launch per timestep;
// h buffers are ping-ponged by the host loop (no intra-grid races).
// grid: (2, H/16), block: 32
// ---------------------------------------------------------------------------
__global__ __launch_bounds__(32)
void gru_step_kernel(const __bf16* __restrict__ hprev_bf, const float* __restrict__ hprev_f,
                     const __bf16* __restrict__ Whh, const float* __restrict__ bhh,
                     const float* __restrict__ gx, int L, int t,
                     float* __restrict__ hnew_f, __bf16* __restrict__ hnew_bf,
                     __bf16* __restrict__ hs_out, int hs_L) {
  const int lane = threadIdx.x;
  const int mbase = blockIdx.x * 16;     // batch tile
  const int jbase = blockIdx.y * 16;     // hidden-column tile
  const int mrow = mbase + (lane & 15);
  const int nr = jbase + (lane & 15);
  v8f cr = {}, cz = {}, cn = {};
  for (int kk = 0; kk < H_; kk += 32) {
    v16bf a = load_a_tile(hprev_bf, H_, mrow, kk, lane);
    v16bf br = load_b_tile(Whh, H_, nr, kk, lane);
    v16bf bz = load_b_tile(Whh, H_, nr + H_, kk, lane);
    v16bf bn = load_b_tile(Whh, H_, nr + 2 * H_, kk, lane);
    cr = WMMA_BF16(a, br, cr);
    cz = WMMA_BF16(a, bz, cz);
    cn = WMMA_BF16(a, bn, cn);
  }
  const int j = jbase + (lane & 15);
  const int mi = mbase + ((lane >> 4) << 3);
  const float bhr = bhh[j];
  const float bhz = bhh[H_ + j];
  const float bhn = bhh[2 * H_ + j];
#pragma unroll
  for (int i = 0; i < 8; ++i) {
    int b = mi + i;
    const float* gxr = gx + ((size_t)b * L + t) * H3_;
    float r = 1.0f / (1.0f + __expf(-(gxr[j] + cr[i] + bhr)));
    float z = 1.0f / (1.0f + __expf(-(gxr[H_ + j] + cz[i] + bhz)));
    float nn = tanhf(gxr[2 * H_ + j] + r * (cn[i] + bhn));
    float hp = hprev_f[(size_t)b * H_ + j];
    float hv = (1.0f - z) * nn + z * hp;
    hnew_f[(size_t)b * H_ + j] = hv;
    hnew_bf[(size_t)b * H_ + j] = (__bf16)hv;
    if (hs_out) hs_out[((size_t)b * hs_L + t) * H_ + j] = (__bf16)hv;
  }
}

// ---------------------------------------------------------------------------
// Output FC: logits = hs[4064,1024] @ fcW[32000,1024]^T + fcb, written to
// out[b, t+1, v] with m = b*127 + t. grid: (254, 500), block: 32
// ---------------------------------------------------------------------------
__device__ __forceinline__ void store_fc_tile(v8f c, int n, int mi,
                                              const float* __restrict__ fcb,
                                              float* __restrict__ out) {
  float bv = fcb[n];
#pragma unroll
  for (int i = 0; i < 8; ++i) {
    int m = mi + i;
    int bb = m / 127;
    int tt = m - bb * 127;
    out[((size_t)(bb * T_ + tt + 1)) * V_ + n] = c[i] + bv;
  }
}

__global__ __launch_bounds__(32)
void fc_kernel(const __bf16* __restrict__ hs, const __bf16* __restrict__ fcW,
               const float* __restrict__ fcb, float* __restrict__ out) {
  const int lane = threadIdx.x;
  const int mbase = blockIdx.x * 16;
  const int nbase = blockIdx.y * 64;
  const int mrow = mbase + (lane & 15);
  const int nr = nbase + (lane & 15);
  v8f c0 = {}, c1 = {}, c2 = {}, c3 = {};
  for (int kk = 0; kk < H_; kk += 32) {
    v16bf a = load_a_tile(hs, H_, mrow, kk, lane);
    v16bf b0 = load_b_tile(fcW, H_, nr, kk, lane);
    v16bf b1 = load_b_tile(fcW, H_, nr + 16, kk, lane);
    v16bf b2 = load_b_tile(fcW, H_, nr + 32, kk, lane);
    v16bf b3 = load_b_tile(fcW, H_, nr + 48, kk, lane);
    c0 = WMMA_BF16(a, b0, c0);
    c1 = WMMA_BF16(a, b1, c1);
    c2 = WMMA_BF16(a, b2, c2);
    c3 = WMMA_BF16(a, b3, c3);
  }
  const int mi = mbase + ((lane >> 4) << 3);
  store_fc_tile(c0, nbase + 0 + (lane & 15), mi, fcb, out);
  store_fc_tile(c1, nbase + 16 + (lane & 15), mi, fcb, out);
  store_fc_tile(c2, nbase + 32 + (lane & 15), mi, fcb, out);
  store_fc_tile(c3, nbase + 48 + (lane & 15), mi, fcb, out);
}

// ---------------------------------------------------------------------------
extern "C" void kernel_launch(void* const* d_in, const int* in_sizes, int n_in,
                              void* d_out, int out_size, void* d_ws, size_t ws_size,
                              hipStream_t stream) {
  (void)in_sizes; (void)n_in; (void)out_size; (void)ws_size;
  const int*   src   = (const int*)d_in[0];
  const int*   tgt   = (const int*)d_in[1];
  const float* emb_e = (const float*)d_in[2];
  const float* Wih_e = (const float*)d_in[3];
  const float* Whh_e = (const float*)d_in[4];
  const float* bih_e = (const float*)d_in[5];
  const float* bhh_e = (const float*)d_in[6];
  const float* emb_d = (const float*)d_in[7];
  const float* Wih_d = (const float*)d_in[8];
  const float* Whh_d = (const float*)d_in[9];
  const float* bih_d = (const float*)d_in[10];
  const float* bhh_d = (const float*)d_in[11];
  const float* fcW   = (const float*)d_in[12];
  const float* fcb   = (const float*)d_in[13];
  float* out = (float*)d_out;

  char* ws = (char*)d_ws;
  size_t off = 0;
  auto alloc = [&](size_t bytes) -> char* {
    char* p = ws + off;
    off = (off + bytes + 255) & ~(size_t)255;
    return p;
  };
  __bf16* wih_e_bf = (__bf16*)alloc((size_t)H3_ * E_ * 2);
  __bf16* whh_e_bf = (__bf16*)alloc((size_t)H3_ * H_ * 2);
  __bf16* wih_d_bf = (__bf16*)alloc((size_t)H3_ * E_ * 2);
  __bf16* whh_d_bf = (__bf16*)alloc((size_t)H3_ * H_ * 2);
  __bf16* fcw_bf   = (__bf16*)alloc((size_t)V_ * H_ * 2);
  __bf16* x_enc    = (__bf16*)alloc((size_t)M_ENC * E_ * 2);
  __bf16* x_dec    = (__bf16*)alloc((size_t)M_DEC * E_ * 2);
  float*  gx_enc   = (float*)alloc((size_t)M_ENC * H3_ * 4);
  float*  gx_dec   = (float*)alloc((size_t)M_DEC * H3_ * 4);
  float*  hf0      = (float*)alloc((size_t)B_ * H_ * 4);
  float*  hf1      = (float*)alloc((size_t)B_ * H_ * 4);
  __bf16* hb0      = (__bf16*)alloc((size_t)B_ * H_ * 2);
  __bf16* hb1      = (__bf16*)alloc((size_t)B_ * H_ * 2);
  __bf16* hs_bf    = (__bf16*)alloc((size_t)M_DEC * H_ * 2);

  // 1) convert weights to bf16 (resident in the 192MB L2 thereafter)
  cvt_bf16_kernel<<<2048, 256, 0, stream>>>(Wih_e, wih_e_bf, (size_t)H3_ * E_);
  cvt_bf16_kernel<<<2048, 256, 0, stream>>>(Whh_e, whh_e_bf, (size_t)H3_ * H_);
  cvt_bf16_kernel<<<2048, 256, 0, stream>>>(Wih_d, wih_d_bf, (size_t)H3_ * E_);
  cvt_bf16_kernel<<<2048, 256, 0, stream>>>(Whh_d, whh_d_bf, (size_t)H3_ * H_);
  cvt_bf16_kernel<<<8192, 256, 0, stream>>>(fcW, fcw_bf, (size_t)V_ * H_);

  // 2) embedding gathers (PAD row zeroed)
  gather_embed_kernel<<<(M_ENC * E_ + 255) / 256, 256, 0, stream>>>(
      src, S_, S_, emb_e, x_enc, M_ENC);
  gather_embed_kernel<<<(M_DEC * E_ + 255) / 256, 256, 0, stream>>>(
      tgt, T_, T_ - 1, emb_d, x_dec, M_DEC);

  // 3) batched input projections gx = X @ Wih^T + bih  (WMMA bf16)
  gemm_xwT_bias_kernel<<<dim3(M_ENC / 16, H3_ / 64), 32, 0, stream>>>(
      x_enc, wih_e_bf, bih_e, gx_enc, E_, H3_);
  gemm_xwT_bias_kernel<<<dim3(M_DEC / 16, H3_ / 64), 32, 0, stream>>>(
      x_dec, wih_d_bf, bih_d, gx_dec, E_, H3_);

  // 4) recurrence: fused gh-GEMM + gating, ping-pong h buffers
  zero_h_kernel<<<(B_ * H_ + 255) / 256, 256, 0, stream>>>(hf0, hb0, B_ * H_);
  float*  hf[2] = {hf0, hf1};
  __bf16* hb[2] = {hb0, hb1};
  int cur = 0;
  for (int t = 0; t < S_; ++t) {  // encoder (hidden only)
    gru_step_kernel<<<dim3(2, H_ / 16), 32, 0, stream>>>(
        hb[cur], hf[cur], whh_e_bf, bhh_e, gx_enc, S_, t,
        hf[1 - cur], hb[1 - cur], (__bf16*)nullptr, 0);
    cur ^= 1;
  }
  for (int t = 0; t < T_ - 1; ++t) {  // decoder (stores hs for FC)
    gru_step_kernel<<<dim3(2, H_ / 16), 32, 0, stream>>>(
        hb[cur], hf[cur], whh_d_bf, bhh_d, gx_dec, T_ - 1, t,
        hf[1 - cur], hb[1 - cur], hs_bf, T_ - 1);
    cur ^= 1;
  }

  // 5) output head: zero t=0 slice, then big WMMA FC into d_out
  zero_t0_kernel<<<(B_ * V_ + 255) / 256, 256, 0, stream>>>(out);
  fc_kernel<<<dim3(M_DEC / 16, V_ / 64), 32, 0, stream>>>(hs_bf, fcw_bf, fcb, out);
}
